// Bilinear_17300128269019
// MI455X (gfx1250) — matrix operand order repaired and checked by reading, not compiled, not run on hardware
//
#include <hip/hip_runtime.h>
#include <stdint.h>

typedef __attribute__((ext_vector_type(16))) __bf16 bf16x16;
typedef __attribute__((ext_vector_type(8)))  float  f32x8;

#define D_DIM  1024
#define N_ROWS 65536
#define T_IDX  65536
#define B_GRP  4096

static __device__ __forceinline__ void split_bf16(float x, __bf16& hi, __bf16& lo) {
    hi = (__bf16)x;                       // RNE truncation to bf16
    lo = (__bf16)(x - (float)hi);         // residual in bf16
}

// ---------------------------------------------------------------------------
// Kernel A: transpose + hi/lo split of W:  W[k][j] (f32) -> WThi/WTlo[j][k] (bf16)
// ---------------------------------------------------------------------------
__global__ void __launch_bounds__(256)
wsplit_kernel(const float* __restrict__ W,
              __bf16* __restrict__ WThi, __bf16* __restrict__ WTlo) {
    __shared__ float tile[32][33];
    const int gx = blockIdx.x;   // j tile
    const int gy = blockIdx.y;   // k tile
    const int x = threadIdx.x, y = threadIdx.y;   // 32 x 8
    #pragma unroll
    for (int i = 0; i < 4; ++i) {
        int k = gy * 32 + y + i * 8;
        int j = gx * 32 + x;
        tile[y + i * 8][x] = W[(size_t)k * D_DIM + j];
    }
    __syncthreads();
    #pragma unroll
    for (int i = 0; i < 4; ++i) {
        int j = gx * 32 + y + i * 8;
        int k = gy * 32 + x;
        float v = tile[x][y + i * 8];
        __bf16 hi, lo; split_bf16(v, hi, lo);
        WThi[(size_t)j * D_DIM + k] = hi;
        WTlo[(size_t)j * D_DIM + k] = lo;
    }
}

// ---------------------------------------------------------------------------
// Kernel B: sal[i] = x_i^T W t_i + b  via split-bf16 WMMA, fused epilogue.
// One WG = 64 rows (4 row-tiles), 8 waves. Wave w owns j-eighth (128 cols),
// processed as 2 groups of 4 j-tiles, computing ALL 4 row-tiles per B fragment
// -> each B fragment feeds 12 WMMAs (4 row-tiles x 3 split terms).
// Per k-iter per wave: 16 global_b128 (B) + 16 ds_b128 (A) -> 48 WMMAs.
// ---------------------------------------------------------------------------
__global__ void __launch_bounds__(256)
saliency_kernel(const float* __restrict__ X, const float* __restrict__ Tm,
                const __bf16* __restrict__ WThi, const __bf16* __restrict__ WTlo,
                const float* __restrict__ bias, float* __restrict__ sal) {
    constexpr int BM = 64;
    constexpr int PITCH = 1032;   // bf16 elems; 2064B row pitch -> conflict-free b128 frag loads
    __shared__ __bf16 sXhi[BM * PITCH];   // 132096 B
    __shared__ __bf16 sXlo[BM * PITCH];   // 132096 B
    __shared__ float  sPart[8][BM][16];   //  32768 B   (total ~297 KB of 320 KB)

    const int tid = threadIdx.x;
    const int r0  = blockIdx.x * BM;

    // Stage + split X[r0 : r0+64][0:1024] into LDS (each element converted once).
    for (int i = tid; i < BM * (D_DIM / 4); i += 256) {
        const int row = i >> 8;            // 256 float4 per row
        const int c4  = (i & 255) * 4;
        const float4 xv = *(const float4*)(X + (size_t)(r0 + row) * D_DIM + c4);
        __bf16 h, l;
        split_bf16(xv.x, h, l); sXhi[row * PITCH + c4 + 0] = h; sXlo[row * PITCH + c4 + 0] = l;
        split_bf16(xv.y, h, l); sXhi[row * PITCH + c4 + 1] = h; sXlo[row * PITCH + c4 + 1] = l;
        split_bf16(xv.z, h, l); sXhi[row * PITCH + c4 + 2] = h; sXlo[row * PITCH + c4 + 2] = l;
        split_bf16(xv.w, h, l); sXhi[row * PITCH + c4 + 3] = h; sXlo[row * PITCH + c4 + 3] = l;
    }
    __syncthreads();

    const int wave = tid >> 5, lane = tid & 31;
    const int l16  = lane & 15, lhi = lane >> 4;

    union F { bf16x16 v; uint4 q[2]; };

    for (int jg = 0; jg < 2; ++jg) {
        const int jbase = wave * 128 + jg * 64;   // 4 j-tiles of 16 cols

        // B fragment lane map (16-bit B 32x16): lane -> N = lane&15, K = (lhi*8 .. +7), (16+lhi*8 .. +7)
        const __bf16* bHiBase = WThi + (size_t)(jbase + l16) * D_DIM + lhi * 8;
        const __bf16* bLoBase = WTlo + (size_t)(jbase + l16) * D_DIM + lhi * 8;

        f32x8 acc[16];                            // [rt*4 + jt]
        #pragma unroll
        for (int t = 0; t < 16; ++t)
            acc[t] = (f32x8){0.f, 0.f, 0.f, 0.f, 0.f, 0.f, 0.f, 0.f};

        #pragma unroll 1                  // keep k a real loop: no hoisting explosion
        for (int k = 0; k < D_DIM; k += 32) {
            // A fragment lane map (16-bit A 16x32): lane -> M = lane&15;
            //   VGPR0-3 = K (lhi*8 .. +7), VGPR4-7 = K (16+lhi*8 .. +7).
            F ah[4], al[4];
            #pragma unroll
            for (int rt = 0; rt < 4; ++rt) {
                const __bf16* ph = sXhi + (rt * 16 + l16) * PITCH + lhi * 8 + k;
                const __bf16* pl = sXlo + (rt * 16 + l16) * PITCH + lhi * 8 + k;
                ah[rt].q[0] = *(const uint4*)(ph);
                ah[rt].q[1] = *(const uint4*)(ph + 16);
                al[rt].q[0] = *(const uint4*)(pl);
                al[rt].q[1] = *(const uint4*)(pl + 16);
            }
            #pragma unroll
            for (int jt = 0; jt < 4; ++jt) {
                const __bf16* bhp = bHiBase + (size_t)jt * 16 * D_DIM + k;
                const __bf16* blp = bLoBase + (size_t)jt * 16 * D_DIM + k;
                F bh, bl;
                bh.q[0] = *(const uint4*)(bhp);
                bh.q[1] = *(const uint4*)(bhp + 16);
                bl.q[0] = *(const uint4*)(blp);
                bl.q[1] = *(const uint4*)(blp + 16);
                #pragma unroll
                for (int rt = 0; rt < 4; ++rt) {
                    // P += Xhi*Whi + Xhi*Wlo + Xlo*Whi   (fp32 accumulate)
                    acc[rt*4+jt] = __builtin_amdgcn_wmma_f32_16x16x32_bf16(false, ah[rt].v, false, bh.v, (short)0, acc[rt*4+jt], false, false);
                    acc[rt*4+jt] = __builtin_amdgcn_wmma_f32_16x16x32_bf16(false, ah[rt].v, false, bl.v, (short)0, acc[rt*4+jt], false, false);
                    acc[rt*4+jt] = __builtin_amdgcn_wmma_f32_16x16x32_bf16(false, al[rt].v, false, bh.v, (short)0, acc[rt*4+jt], false, false);
                }
            }
        }

        // Fused epilogue: partial_sal[row] += sum_j P[row][j] * T[row][j].
        // C/D layout: lane holds (M = v + 8*lhi, N = l16). Accumulate in LDS
        // (same-wave, same-address RMW -> in-order, deterministic).
        #pragma unroll
        for (int rt = 0; rt < 4; ++rt) {
            #pragma unroll
            for (int v = 0; v < 8; ++v) {
                const int row = rt * 16 + v + 8 * lhi;
                float s = 0.f;
                #pragma unroll
                for (int jt = 0; jt < 4; ++jt)
                    s += acc[rt*4+jt][v] *
                         Tm[(size_t)(r0 + row) * D_DIM + jbase + jt * 16 + l16];
                if (jg == 0) sPart[wave][row][l16] = s;
                else         sPart[wave][row][l16] += s;
            }
        }
    }

    __syncthreads();
    if (tid < BM) {
        float s = *bias;
        #pragma unroll
        for (int q = 0; q < 8; ++q)
            for (int c = 0; c < 16; ++c)
                s += sPart[q][tid][c];
        sal[r0 + tid] = s;
    }
}

// ---------------------------------------------------------------------------
// Kernel C: group start offsets from sorted segment_ids.
// ---------------------------------------------------------------------------
__global__ void gstart_kernel(const int* __restrict__ seg, int* __restrict__ gstart) {
    const int t = blockIdx.x * 256 + threadIdx.x;
    if (t >= T_IDX) return;
    if (t == 0) gstart[0] = 0;
    else if (seg[t] != seg[t - 1]) gstart[seg[t]] = t;
    if (t == T_IDX - 1) gstart[B_GRP] = T_IDX;
}

// ---------------------------------------------------------------------------
// Kernel D: per-group segment softmax + weighted template sum + concat output.
// One 256-thread WG per group; thread t owns columns 4t..4t+3.
// ---------------------------------------------------------------------------
__global__ void __launch_bounds__(256)
softmax_templ_kernel(const float* __restrict__ Xin, const float* __restrict__ Tm,
                     const float* __restrict__ sal,
                     const int* __restrict__ flat_idx, const int* __restrict__ first_idx,
                     const int* __restrict__ gstart,
                     float* __restrict__ outMain, float* __restrict__ outLogits) {
    __shared__ float red[256];
    const int b = blockIdx.x, tid = threadIdx.x;
    const int s = gstart[b], e = gstart[b + 1];

    // pass 1: group max
    float m = -3.402823466e+38f;
    for (int i = s + tid; i < e; i += 256) m = fmaxf(m, sal[flat_idx[i]]);
    red[tid] = m; __syncthreads();
    for (int st = 128; st > 0; st >>= 1) {
        if (tid < st) red[tid] = fmaxf(red[tid], red[tid + st]);
        __syncthreads();
    }
    const float mm = red[0];
    __syncthreads();

    // pass 2: sum of exp; emit raw logits (second output)
    float ssum = 0.f;
    for (int i = s + tid; i < e; i += 256) {
        const float l = sal[flat_idx[i]];
        outLogits[i] = l;
        ssum += __expf(l - mm);
    }
    red[tid] = ssum; __syncthreads();
    for (int st = 128; st > 0; st >>= 1) {
        if (tid < st) red[tid] += red[tid + st];
        __syncthreads();
    }
    const float inv = 1.f / red[0];

    // pass 3: weighted template sum (fixed order -> deterministic)
    float4 acc = {0.f, 0.f, 0.f, 0.f};
    const int c = tid * 4;
    for (int i = s; i < e; ++i) {
        const int idx = flat_idx[i];
        const float w = __expf(sal[idx] - mm) * inv;
        const float4 xv = *(const float4*)(Tm + (size_t)idx * D_DIM + c);
        acc.x += w * xv.x; acc.y += w * xv.y; acc.z += w * xv.z; acc.w += w * xv.w;
    }

    float* outRow = outMain + (size_t)b * (2 * D_DIM);
    const int fi = first_idx[b];
    const float4 xin = *(const float4*)(Xin + (size_t)fi * D_DIM + c);
    *(float4*)(outRow + c) = xin;
    *(float4*)(outRow + D_DIM + c) = acc;
}

// ---------------------------------------------------------------------------
extern "C" void kernel_launch(void* const* d_in, const int* in_sizes, int n_in,
                              void* d_out, int out_size, void* d_ws, size_t ws_size,
                              hipStream_t stream) {
    const float* x_input     = (const float*)d_in[0];
    const float* x_templates = (const float*)d_in[1];
    const float* W           = (const float*)d_in[2];
    const float* bias        = (const float*)d_in[3];
    const int*   flat_idx    = (const int*)d_in[4];
    const int*   seg         = (const int*)d_in[5];
    const int*   first_idx   = (const int*)d_in[6];

    char* ws = (char*)d_ws;
    __bf16* WThi  = (__bf16*)ws;                                            // 2 MB
    __bf16* WTlo  = (__bf16*)(ws + (size_t)D_DIM * D_DIM * 2);              // 2 MB
    float*  sal   = (float*)(ws + (size_t)D_DIM * D_DIM * 4);               // 256 KB
    int*    gstart = (int*)(ws + (size_t)D_DIM * D_DIM * 4 + (size_t)N_ROWS * 4);

    float* outMain   = (float*)d_out;
    float* outLogits = outMain + (size_t)B_GRP * 2 * D_DIM;

    wsplit_kernel<<<dim3(D_DIM / 32, D_DIM / 32), dim3(32, 8), 0, stream>>>(W, WThi, WTlo);
    saliency_kernel<<<N_ROWS / 64, 256, 0, stream>>>(x_input, x_templates, WThi, WTlo, bias, sal);
    gstart_kernel<<<T_IDX / 256, 256, 0, stream>>>(seg, gstart);
    softmax_templ_kernel<<<B_GRP, 256, 0, stream>>>(x_input, x_templates, sal,
                                                    flat_idx, first_idx, gstart,
                                                    outMain, outLogits);
}
